// TransformersLayer_87239375716863
// MI455X (gfx1250) — matrix operand order repaired and checked by reading, not compiled, not run on hardware
//
#include <hip/hip_runtime.h>
#include <hip/hip_bf16.h>

// ---------------------------------------------------------------------------
// Types for CDNA5 WMMA (wave32): v_wmma_f32_16x16x32_bf16
// ---------------------------------------------------------------------------
typedef __bf16 bf16;
typedef __attribute__((ext_vector_type(16))) __bf16 v16bf;
typedef __attribute__((ext_vector_type(8)))  __bf16 v8bf;
typedef __attribute__((ext_vector_type(8)))  float  v8f;
typedef __attribute__((ext_vector_type(4)))  int    v4i;

#define AS1 __attribute__((address_space(1)))
#define AS3 __attribute__((address_space(3)))

static constexpr int S_ = 2048, B_ = 2, D_ = 1024, H_ = 16, F_ = 4096;
static constexpr int M_ = S_ * B_;   // 4096 flattened (s,b) rows
static constexpr int DK_ = D_ / H_;  // 64

// ---- CDNA5 async global->LDS (probe-confirmed on this toolchain) ----------
#if defined(__has_builtin)
#  if __has_builtin(__builtin_amdgcn_global_load_async_to_lds_b128)
#    define USE_ASYNC_LDS 1
#  endif
#  if __has_builtin(__builtin_amdgcn_s_wait_asynccnt)
#    define HAVE_WAIT_ASYNC 1
#  endif
#endif

__device__ __forceinline__ void copy16_g2l(const bf16* g, bf16* l) {
#if defined(USE_ASYNC_LDS)
  __builtin_amdgcn_global_load_async_to_lds_b128(
      (AS1 v4i*)g, (AS3 v4i*)l, 0, 0);
#else
  *(v8bf*)l = *(const v8bf*)g;
#endif
}

#if defined(USE_ASYNC_LDS) && defined(HAVE_WAIT_ASYNC)
#  define WAIT_ASYNC(n) __builtin_amdgcn_s_wait_asynccnt(n)
#else
#  define WAIT_ASYNC(n) ((void)0)
#endif

__device__ __forceinline__ v8f zero8() {
  v8f z = {0.f, 0.f, 0.f, 0.f, 0.f, 0.f, 0.f, 0.f};
  return z;
}

// A-matrix fragment (16x32 bf16): lane holds row (lane&15); K contiguous:
// lanes<16 -> K 0..15, lanes>=16 -> K 16..31.
__device__ __forceinline__ v16bf load_frag_a(const bf16* p) {
  v8bf lo = *(const v8bf*)p;
  v8bf hi = *(const v8bf*)(p + 8);
  v16bf r;
#pragma unroll
  for (int e = 0; e < 8; ++e) { r[e] = lo[e]; r[e + 8] = hi[e]; }
  return r;
}

// B-matrix fragment (32x16 bf16): lane holds column (lane&15); K contiguous:
// lanes<16 -> K 0..15, lanes>=16 -> K 16..31. Caller passes p = col_ptr+lhi*16.
__device__ __forceinline__ v16bf load_frag_b(const bf16* p) {
  v8bf lo = *(const v8bf*)p;
  v8bf hi = *(const v8bf*)(p + 8);
  v16bf r;
#pragma unroll
  for (int e = 0; e < 8; ++e) { r[e] = lo[e]; r[e + 8] = hi[e]; }
  return r;
}

__device__ __forceinline__ v8f wmma_bf16(v16bf a, v16bf b, v8f c) {
  return __builtin_amdgcn_wmma_f32_16x16x32_bf16(
      false, a, false, b, (short)0, c, false, false);
}

// ---------------------------------------------------------------------------
// fp32 -> bf16 convert (weights)
// ---------------------------------------------------------------------------
__global__ __launch_bounds__(256) void cvt_bf16_kernel(
    const float* __restrict__ in, bf16* __restrict__ out, int n) {
  int i = blockIdx.x * 256 + threadIdx.x;
  if (i < n) out[i] = (bf16)in[i];
}

// ---------------------------------------------------------------------------
// LayerNorm over D=1024, one block per row, bf16 output for the GEMMs.
// ---------------------------------------------------------------------------
__global__ __launch_bounds__(256) void layernorm_bf16_kernel(
    const float* __restrict__ x, const float* __restrict__ g,
    const float* __restrict__ b, bf16* __restrict__ y) {
  __shared__ float rs_[256], rss_[256];
  const int row = blockIdx.x;
  const float* xr = x + (size_t)row * D_;
  float s = 0.f, ss = 0.f;
#pragma unroll
  for (int i = threadIdx.x; i < D_; i += 256) {
    float v = xr[i];
    s += v; ss += v * v;
  }
  rs_[threadIdx.x] = s; rss_[threadIdx.x] = ss;
  __syncthreads();
  for (int off = 128; off > 0; off >>= 1) {
    if (threadIdx.x < off) {
      rs_[threadIdx.x]  += rs_[threadIdx.x + off];
      rss_[threadIdx.x] += rss_[threadIdx.x + off];
    }
    __syncthreads();
  }
  const float mu  = rs_[0] * (1.f / D_);
  const float var = rss_[0] * (1.f / D_) - mu * mu;
  const float inv = rsqrtf(var + 1e-5f);
  bf16* yr = y + (size_t)row * D_;
#pragma unroll
  for (int i = threadIdx.x; i < D_; i += 256)
    yr[i] = (bf16)((xr[i] - mu) * inv * g[i] + b[i]);
}

// ---------------------------------------------------------------------------
// Tiled WMMA GEMM:  C[M,N] = A[M,K] @ W[N,K]^T  (+bias)(+relu)(+resid)
// Block = 256 threads = 8 waves; 128x128 tile, waves 4x2, each owning 32x64
// (2x4 WMMA 16x16 tiles, 8 WMMA per K=32 step). Double-buffered LDS with
// async global->LDS copies (ASYNCcnt).
// CbT (optional) additionally writes the result transposed as
// [b][col][s] with row = s*B_+b (used by the QKV GEMM so attention can read
// V columns with contiguous vector loads instead of 2-byte gathers).
// Assumes M,N multiples of 128 and K multiple of 32 (true for all calls).
// ---------------------------------------------------------------------------
#define GTM 128
#define GTN 128
#define GTK 32
#define LDSK (GTK + 8)   // 40 bf16 = 80B row pitch: conflict-free frag reads

__global__ __launch_bounds__(256) void gemm_bf16_kernel(
    const bf16* __restrict__ A, const bf16* __restrict__ W,
    const float* __restrict__ bias, const float* __restrict__ resid,
    float* __restrict__ Cf, bf16* __restrict__ Cb, bf16* __restrict__ CbT,
    int M, int N, int K, int fuse_relu) {
  __shared__ bf16 sA[2][GTM * LDSK];
  __shared__ bf16 sB[2][GTN * LDSK];

  const int tid  = threadIdx.x;
  const int bm   = blockIdx.x * GTM;
  const int bn   = blockIdx.y * GTN;
  const int wave = tid >> 5;          // 0..7
  const int lane = tid & 31;
  const int lrow = lane & 15;
  const int lhi  = lane >> 4;
  const int wm   = (wave >> 1) * 32;  // 0,32,64,96
  const int wn   = (wave & 1) * 64;   // 0,64

  // Stage loaders: 256 threads, each moves 32B (two b128) per tile.
  const int ldRow  = tid >> 1;        // 0..127
  const int ldColE = (tid & 1) * 16;  // element offset 0 or 16

  const bf16* gA = A + (size_t)(bm + ldRow) * K + ldColE;
  const bf16* gW = W + (size_t)(bn + ldRow) * K + ldColE;

  v8f acc[2][4];
#pragma unroll
  for (int i = 0; i < 2; ++i)
#pragma unroll
    for (int j = 0; j < 4; ++j) acc[i][j] = zero8();

  // ---- stage 0 ----
  {
    bf16* la = &sA[0][ldRow * LDSK + ldColE];
    bf16* lb = &sB[0][ldRow * LDSK + ldColE];
    copy16_g2l(gA, la);       copy16_g2l(gA + 8, la + 8);
    copy16_g2l(gW, lb);       copy16_g2l(gW + 8, lb + 8);
  }

  for (int k0 = 0; k0 < K; k0 += GTK) {
    const int buf  = (k0 / GTK) & 1;
    const bool more = (k0 + GTK) < K;
    if (more) {   // issue next stage while current computes
      const bf16* na = gA + k0 + GTK;
      const bf16* nw = gW + k0 + GTK;
      bf16* la = &sA[buf ^ 1][ldRow * LDSK + ldColE];
      bf16* lb = &sB[buf ^ 1][ldRow * LDSK + ldColE];
      copy16_g2l(na, la);     copy16_g2l(na + 8, la + 8);
      copy16_g2l(nw, lb);     copy16_g2l(nw + 8, lb + 8);
      __builtin_prefetch(na + GTK, 0, 0);   // global_prefetch_b8
      __builtin_prefetch(nw + GTK, 0, 0);
      WAIT_ASYNC(4);          // my 4 current-stage copies have landed
    } else {
      WAIT_ASYNC(0);
    }
    __syncthreads();

    v16bf af[2], bfg[4];
#pragma unroll
    for (int i = 0; i < 2; ++i)
      af[i] = load_frag_a(&sA[buf][(wm + 16 * i + lrow) * LDSK + lhi * 16]);
#pragma unroll
    for (int j = 0; j < 4; ++j)
      bfg[j] = load_frag_b(&sB[buf][(wn + 16 * j + lrow) * LDSK + lhi * 16]);

#pragma unroll
    for (int i = 0; i < 2; ++i)
#pragma unroll
      for (int j = 0; j < 4; ++j)
        acc[i][j] = wmma_bf16(af[i], bfg[j], acc[i][j]);
    __syncthreads();
  }

  // Epilogue. C layout: lane col = lane&15; rows = (lane>=16 ? 8..15 : 0..7).
#pragma unroll
  for (int i = 0; i < 2; ++i) {
#pragma unroll
    for (int j = 0; j < 4; ++j) {
      const int col = bn + wn + 16 * j + lrow;
      const float bv = bias ? bias[col] : 0.f;
#pragma unroll
      for (int r = 0; r < 8; ++r) {
        const int row = bm + wm + 16 * i + lhi * 8 + r;
        float v = acc[i][j][r] + bv;
        if (fuse_relu) v = v > 0.f ? v : 0.f;
        if (resid) v += resid[(size_t)row * N + col];
        const size_t idx = (size_t)row * N + col;
        if (Cf) Cf[idx] = v;
        if (Cb) Cb[idx] = (bf16)v;
        if (CbT) {   // transposed copy: [b][col][s], row = s*B_+b
          const int s = row / B_, bb = row % B_;
          CbT[((size_t)bb * N + col) * S_ + s] = (bf16)v;
        }
      }
    }
  }
}

// ---------------------------------------------------------------------------
// Fused attention with softmax over the HEADS axis (faithful to reference).
// qkv (q=k=v) bf16 [S,B,D]; qkvT bf16 [B,D,S] (transposed copy from the QKV
// GEMM epilogue). One block per (16 q-rows, b): 256 thr = 8 waves, wave w
// owns heads {2w, 2w+1}. Stream keys 32 at a time:
//   score tile via WMMA (Q 16xDK . K^T) -> LDS [h][16q][32k]
//   softmax across the 16 heads per (q,k) pair (local! no cross-k state)
//   PV via WMMA (attn 16x32 . V 32x16-per-dtile) with V fragments read as
//   contiguous vectors from qkvT -- no 2-byte gathers.
// No [S,S,B,H] tensor ever touches HBM (reference would need 537MB for it).
// ---------------------------------------------------------------------------
__global__ __launch_bounds__(256) void attention_headsm_kernel(
    const bf16* __restrict__ qkv, const bf16* __restrict__ qkvT,
    bf16* __restrict__ outb) {
  __shared__ float sc[H_][16][32];   // 32 KB scores->attn

  const int q0   = blockIdx.x * 16;
  const int b    = blockIdx.y;
  const int tid  = threadIdx.x;
  const int wave = tid >> 5;
  const int lane = tid & 31;
  const int lrow = lane & 15;
  const int lhi  = lane >> 4;
  const float scale = 0.125f;        // 1/sqrt(DK=64)

  // Preload Q A-fragments (per wave: 2 heads x 2 K-chunks of 32)
  v16bf qf[2][2];
#pragma unroll
  for (int hh = 0; hh < 2; ++hh) {
    const int h = wave * 2 + hh;
    const bf16* qp = qkv + ((size_t)(q0 + lrow) * B_ + b) * D_ + h * DK_;
#pragma unroll
    for (int c = 0; c < 2; ++c)
      qf[hh][c] = load_frag_a(qp + c * 32 + lhi * 16);
  }

  v8f o[2][4];
#pragma unroll
  for (int hh = 0; hh < 2; ++hh)
#pragma unroll
    for (int j = 0; j < 4; ++j) o[hh][j] = zero8();

  for (int k0 = 0; k0 < S_; k0 += 32) {
    // ---- scores: per head, two 16-wide key tiles, K=64 over Q*K^T ----
#pragma unroll
    for (int hh = 0; hh < 2; ++hh) {
      const int h = wave * 2 + hh;
#pragma unroll
      for (int kt = 0; kt < 2; ++kt) {
        v8f s = zero8();
        const bf16* kp =
            qkv + ((size_t)(k0 + kt * 16 + lrow) * B_ + b) * D_ + h * DK_;
#pragma unroll
        for (int c = 0; c < 2; ++c) {
          v16bf bk = load_frag_b(kp + c * 32 + lhi * 16);
          s = wmma_bf16(qf[hh][c], bk, s);
        }
#pragma unroll
        for (int r = 0; r < 8; ++r)
          sc[h][lhi * 8 + r][kt * 16 + lrow] = s[r] * scale;
      }
    }
    __syncthreads();

    // ---- softmax across heads: 512 (q,k) pairs, 2 per thread ----
    for (int p = tid; p < 16 * 32; p += 256) {
      const int qq = p >> 5, kk = p & 31;
      float m = -INFINITY;
#pragma unroll
      for (int h = 0; h < H_; ++h) m = fmaxf(m, sc[h][qq][kk]);
      float e[H_]; float sum = 0.f;
#pragma unroll
      for (int h = 0; h < H_; ++h) { e[h] = __expf(sc[h][qq][kk] - m); sum += e[h]; }
      const float inv = 1.f / sum;
#pragma unroll
      for (int h = 0; h < H_; ++h) sc[h][qq][kk] = e[h] * inv;
    }
    __syncthreads();

    // ---- PV: out[16q x 64d] += attn[16x32] @ V[32 x 64] per head ----
#pragma unroll
    for (int hh = 0; hh < 2; ++hh) {
      const int h = wave * 2 + hh;
      v16bf afr;
      const int kb0 = lhi * 16;
#pragma unroll
      for (int e = 0; e < 8; ++e) {
        afr[e]     = (bf16)sc[h][lrow][kb0 + e];
        afr[e + 8] = (bf16)sc[h][lrow][kb0 + 8 + e];
      }
#pragma unroll
      for (int j = 0; j < 4; ++j) {
        // V fragment from transposed copy: lane col = d, keys contiguous.
        const size_t dcol = (size_t)h * DK_ + j * 16 + lrow;
        const bf16* vp = qkvT + ((size_t)b * D_ + dcol) * S_ + k0;
        v16bf bfr = load_frag_b(vp + lhi * 16);
        o[hh][j] = wmma_bf16(afr, bfr, o[hh][j]);
      }
    }
    __syncthreads();
  }

  // ---- store [16q x D] bf16 ----
#pragma unroll
  for (int hh = 0; hh < 2; ++hh) {
    const int h = wave * 2 + hh;
#pragma unroll
    for (int j = 0; j < 4; ++j) {
      const int d = h * DK_ + j * 16 + lrow;
#pragma unroll
      for (int r = 0; r < 8; ++r) {
        const int q = q0 + lhi * 8 + r;
        outb[((size_t)q * B_ + b) * D_ + d] = (bf16)o[hh][j][r];
      }
    }
  }
}

// ---------------------------------------------------------------------------
// Host-side pipeline
// ---------------------------------------------------------------------------
extern "C" void kernel_launch(void* const* d_in, const int* in_sizes, int n_in,
                              void* d_out, int out_size, void* d_ws, size_t ws_size,
                              hipStream_t stream) {
  const float* x     = (const float*)d_in[0];
  const float* ln1_g = (const float*)d_in[1];
  const float* ln1_b = (const float*)d_in[2];
  const float* ln2_g = (const float*)d_in[3];
  const float* ln2_b = (const float*)d_in[4];
  const float* Ws    = (const float*)d_in[5];
  const float* bs    = (const float*)d_in[6];
  const float* Wo    = (const float*)d_in[7];
  const float* bo    = (const float*)d_in[8];
  const float* W1    = (const float*)d_in[9];
  const float* b1    = (const float*)d_in[10];
  const float* W2    = (const float*)d_in[11];
  const float* b2    = (const float*)d_in[12];
  float* out = (float*)d_out;

  size_t off = 0;
  auto alloc = [&](size_t bytes) -> void* {
    void* p = (char*)d_ws + off;
    off = (off + bytes + 255) & ~(size_t)255;
    return p;
  };
  bf16* Wsb   = (bf16*)alloc((size_t)D_ * D_ * 2);
  bf16* Wob   = (bf16*)alloc((size_t)D_ * D_ * 2);
  bf16* W1b   = (bf16*)alloc((size_t)F_ * D_ * 2);
  bf16* W2b   = (bf16*)alloc((size_t)D_ * F_ * 2);
  bf16* xln1  = (bf16*)alloc((size_t)M_ * D_ * 2);
  bf16* qkvb  = (bf16*)alloc((size_t)M_ * D_ * 2);
  bf16* qkvT  = (bf16*)alloc((size_t)M_ * D_ * 2);   // [B][D][S]
  bf16* attnb = (bf16*)alloc((size_t)M_ * D_ * 2);
  float* x2   = (float*)alloc((size_t)M_ * D_ * 4);
  bf16* xln2  = (bf16*)alloc((size_t)M_ * D_ * 2);
  bf16* hbuf  = (bf16*)alloc((size_t)M_ * F_ * 2);
  (void)ws_size; (void)n_in; (void)in_sizes; (void)out_size;

  // weights -> bf16
  cvt_bf16_kernel<<<(D_ * D_) / 256, 256, 0, stream>>>(Ws, Wsb, D_ * D_);
  cvt_bf16_kernel<<<(D_ * D_) / 256, 256, 0, stream>>>(Wo, Wob, D_ * D_);
  cvt_bf16_kernel<<<(F_ * D_) / 256, 256, 0, stream>>>(W1, W1b, F_ * D_);
  cvt_bf16_kernel<<<(D_ * F_) / 256, 256, 0, stream>>>(W2, W2b, D_ * F_);

  // LN1
  layernorm_bf16_kernel<<<M_, 256, 0, stream>>>(x, ln1_g, ln1_b, xln1);

  // qkv = LN1(x) @ Ws^T + bs  (bf16 out, + transposed copy for V access)
  gemm_bf16_kernel<<<dim3(M_ / GTM, D_ / GTN), 256, 0, stream>>>(
      xln1, Wsb, bs, nullptr, nullptr, qkvb, qkvT, M_, D_, D_, 0);

  // fused attention (softmax over heads), bf16 out
  attention_headsm_kernel<<<dim3(S_ / 16, B_), 256, 0, stream>>>(
      qkvb, qkvT, attnb);

  // x2 = attn @ Wo^T + bo + x  (fp32 out)
  gemm_bf16_kernel<<<dim3(M_ / GTM, D_ / GTN), 256, 0, stream>>>(
      attnb, Wob, bo, x, x2, nullptr, nullptr, M_, D_, D_, 0);

  // LN2
  layernorm_bf16_kernel<<<M_, 256, 0, stream>>>(x2, ln2_g, ln2_b, xln2);

  // h = relu(LN2 @ W1^T + b1)  (bf16 out)
  gemm_bf16_kernel<<<dim3(M_ / GTM, F_ / GTN), 256, 0, stream>>>(
      xln2, W1b, b1, nullptr, nullptr, hbuf, nullptr, M_, F_, D_, 1);

  // out = h @ W2^T + b2 + x2  (fp32 out)
  gemm_bf16_kernel<<<dim3(M_ / GTM, D_ / GTN), 256, 0, stream>>>(
      hbuf, W2b, b2, x2, out, nullptr, nullptr, M_, D_, F_, 0);
}
